// IxformerQuantMoe_42889543418235
// MI455X (gfx1250) — 8-bit matrix operand order repaired and checked by reading, compiled, not run on hardware
//
#include <hip/hip_runtime.h>
#include <hip/hip_bf16.h>
#include <stdint.h>

#define T_TOK 4096
#define H_DIM 2048
#define I_DIM 1408
#define E_EXP 8
#define TWO_I 2816

typedef int v8i __attribute__((ext_vector_type(8)));

__global__ __launch_bounds__(128) void gate_kernel(const float* __restrict__ hs,
                                                   const float* __restrict__ gw,
                                                   float* __restrict__ route) {
    const int wave = threadIdx.x >> 5;
    const int lane = threadIdx.x & 31;
    const int t = blockIdx.x * 4 + wave;
    if (t >= T_TOK) return;

    float acc[E_EXP];
#pragma unroll
    for (int e = 0; e < E_EXP; ++e) acc[e] = 0.f;

    for (int h = lane; h < H_DIM; h += 32) {
        const float x = hs[(size_t)t * H_DIM + h];
#pragma unroll
        for (int e = 0; e < E_EXP; ++e) acc[e] += x * gw[e * H_DIM + h];
    }
#pragma unroll
    for (int e = 0; e < E_EXP; ++e) {
#pragma unroll
        for (int off = 16; off > 0; off >>= 1) acc[e] += __shfl_xor(acc[e], off, 32);
    }
    if (lane == 0) {
        int i1 = 0;
#pragma unroll
        for (int e = 1; e < E_EXP; ++e) if (acc[e] > acc[i1]) i1 = e;
        int i2 = (i1 == 0) ? 1 : 0;
#pragma unroll
        for (int e = 0; e < E_EXP; ++e) if (e != i1 && acc[e] > acc[i2]) i2 = e;
        const float g1 = 1.f / (1.f + __expf(acc[i2] - acc[i1]));
#pragma unroll
        for (int e = 0; e < E_EXP; ++e)
            route[(size_t)t * E_EXP + e] = (e == i1) ? g1 : ((e == i2) ? (1.f - g1) : 0.f);
    }
}

__global__ __launch_bounds__(256) void quant_kernel(const float* __restrict__ hs,
                                                    int8_t* __restrict__ q,
                                                    float* __restrict__ s) {
    __shared__ float red[256];
    const int t = blockIdx.x;
    const int tid = threadIdx.x;
    float m = 0.f;
    for (int h = tid; h < H_DIM; h += 256) m = fmaxf(m, fabsf(hs[(size_t)t * H_DIM + h]));
    red[tid] = m;
    __syncthreads();
    for (int o = 128; o > 0; o >>= 1) {
        if (tid < o) red[tid] = fmaxf(red[tid], red[tid + o]);
        __syncthreads();
    }
    const float sc = fmaxf(red[0] * (1.f / 127.f), 1e-8f);
    if (tid == 0) s[t] = sc;
    const float inv = 1.f / sc;
    for (int h = tid; h < H_DIM; h += 256) {
        float v = rintf(hs[(size_t)t * H_DIM + h] * inv);
        v = fminf(fmaxf(v, -127.f), 127.f);
        q[(size_t)t * H_DIM + h] = (int8_t)v;
    }
}

#define SM_QA   0
#define SM_ACTF 32768
#define SM_QACT (32768 + 90112)
#define SM_RG   (SM_QACT + 22528)
#define SM_SIN  (SM_RG + 64)
#define SM_RMX  (SM_SIN + 64)
#define SM_SACT (SM_RMX + 64)
#define SM_FLAG (SM_SACT + 64)
#define SM_TOTAL (SM_FLAG + 16)

__device__ __forceinline__ v8i load_a_frag(const int8_t* ap) {
    v8i a;
    ((int4*)&a)[0] = *(const int4*)(ap + 0);
    ((int4*)&a)[1] = *(const int4*)(ap + 32);
    return a;
}
__device__ __forceinline__ v8i load_b_frag(const int8_t* bp) {
    v8i b;
    ((int4*)&b)[0] = *(const int4*)(bp + 0);
    ((int4*)&b)[1] = *(const int4*)(bp + 32);
    return b;
}

__global__ __launch_bounds__(256) void moe_kernel(const int8_t* __restrict__ q_in,
                                                  const float* __restrict__ s_in,
                                                  const float* __restrict__ route,
                                                  const int8_t* __restrict__ w13,
                                                  const float* __restrict__ s13,
                                                  const int8_t* __restrict__ w2,
                                                  const float* __restrict__ s2w,
                                                  float* __restrict__ out) {
    extern __shared__ char smem[];
    int8_t* qA   = (int8_t*)(smem + SM_QA);
    float*  actF = (float*)(smem + SM_ACTF);
    int8_t* qAct = (int8_t*)(smem + SM_QACT);
    float*  rgate = (float*)(smem + SM_RG);
    float*  sins  = (float*)(smem + SM_SIN);
    int*    rmx   = (int*)(smem + SM_RMX);
    float*  sact  = (float*)(smem + SM_SACT);
    int*    sflag = (int*)(smem + SM_FLAG);

    const int tid  = threadIdx.x;
    const int wave = tid >> 5;
    const int lane = tid & 31;
    const int n    = lane & 15;
    const int hh   = lane >> 4;
    const int t0   = blockIdx.x * 16;

    for (int idx = tid; idx < 16 * H_DIM; idx += 256) {
        const int m = idx >> 11, c = idx & (H_DIM - 1);
        out[(size_t)(t0 + m) * H_DIM + c] = 0.f;
        qA[idx] = q_in[(size_t)(t0 + m) * H_DIM + c];
    }
    if (tid < 16) sins[tid] = s_in[t0 + tid];
    __syncthreads();

    float smr[8];
#pragma unroll
    for (int r = 0; r < 8; ++r) smr[r] = sins[hh * 8 + r];

    for (int e = 0; e < E_EXP; ++e) {
        if (tid < 16) { rgate[tid] = route[(size_t)(t0 + tid) * E_EXP + e]; rmx[tid] = 0; }
        if (tid == 0) *sflag = 0;
        __syncthreads();
        if (tid < 16 && rgate[tid] != 0.f) atomicOr(sflag, 1);
        __syncthreads();
        if (*sflag == 0) continue;

        const int8_t* w13e = w13 + (size_t)e * TWO_I * H_DIM;
        const float*  s13e = s13 + e * TWO_I;

        float lmax[8];
#pragma unroll
        for (int r = 0; r < 8; ++r) lmax[r] = 0.f;

        for (int p = 0; p < 11; ++p) {
            const int j  = wave * 11 + p;
            const int cg = j * 16 + n;
            const int cu = cg + I_DIM;
            const int8_t* bgp = w13e + (size_t)cg * H_DIM + hh * 16;
            const int8_t* bup = w13e + (size_t)cu * H_DIM + hh * 16;
            const int8_t* ap0 = qA + ((size_t)n << 11) + hh * 16;
            v8i accg = {}, accu = {};
            for (int kk = 0; kk < H_DIM; kk += 64) {
                const v8i a  = load_a_frag(ap0 + kk);
                const v8i bg = load_b_frag(bgp + kk);
                const v8i bu = load_b_frag(bup + kk);
                __builtin_prefetch(bgp + kk + 128, 0, 1);
                __builtin_prefetch(bup + kk + 128, 0, 1);
                accg = __builtin_amdgcn_wmma_i32_16x16x64_iu8(true, a, true, bg, accg, false, false);
                accu = __builtin_amdgcn_wmma_i32_16x16x64_iu8(true, a, true, bu, accu, false, false);
            }
            const float sg = s13e[cg];
            const float su = s13e[cu];
#pragma unroll
            for (int r = 0; r < 8; ++r) {
                const float g = (float)accg[r] * smr[r] * sg;
                const float u = (float)accu[r] * smr[r] * su;
                const float act = (g / (1.f + __expf(-g))) * u;
                actF[(hh * 8 + r) * I_DIM + j * 16 + n] = act;
                lmax[r] = fmaxf(lmax[r], fabsf(act));
            }
        }
#pragma unroll
        for (int r = 0; r < 8; ++r)
            atomicMax(&rmx[hh * 8 + r], __float_as_int(lmax[r]));
        __syncthreads();
        if (tid < 16) sact[tid] = fmaxf(__int_as_float(rmx[tid]) * (1.f / 127.f), 1e-8f);
        __syncthreads();

        for (int idx = tid; idx < 16 * I_DIM; idx += 256) {
            const int m = idx / I_DIM;
            float v = rintf(actF[idx] / sact[m]);
            v = fminf(fmaxf(v, -127.f), 127.f);
            qAct[idx] = (int8_t)v;
        }
        __syncthreads();

        float rgr[8], sar[8];
#pragma unroll
        for (int r = 0; r < 8; ++r) { rgr[r] = rgate[hh * 8 + r]; sar[r] = sact[hh * 8 + r]; }

        const int8_t* w2e = w2 + (size_t)e * H_DIM * I_DIM;
        const float*  s2e = s2w + e * H_DIM;
        const int8_t* ap0 = qAct + (size_t)n * I_DIM + hh * 16;
        for (int qn = 0; qn < 16; ++qn) {
            const int cn = (wave * 16 + qn) * 16 + n;
            const int8_t* bp = w2e + (size_t)cn * I_DIM + hh * 16;
            v8i acc = {};
            for (int kk = 0; kk < I_DIM; kk += 64) {
                const v8i a = load_a_frag(ap0 + kk);
                const v8i b = load_b_frag(bp + kk);
                __builtin_prefetch(bp + kk + 128, 0, 1);
                acc = __builtin_amdgcn_wmma_i32_16x16x64_iu8(true, a, true, b, acc, false, false);
            }
            const float sw = s2e[cn];
#pragma unroll
            for (int r = 0; r < 8; ++r) {
                const size_t o = (size_t)(t0 + hh * 8 + r) * H_DIM + cn;
                out[o] += rgr[r] * sar[r] * (float)acc[r] * sw;
            }
        }
        __syncthreads();
    }
}

extern "C" void kernel_launch(void* const* d_in, const int* in_sizes, int n_in,
                              void* d_out, int out_size, void* d_ws, size_t ws_size,
                              hipStream_t stream) {
    const float*  hs  = (const float*)d_in[0];
    const float*  gw  = (const float*)d_in[1];
    const int8_t* w13 = (const int8_t*)d_in[2];
    const float*  s13 = (const float*)d_in[3];
    const int8_t* w2  = (const int8_t*)d_in[4];
    const float*  s2w = (const float*)d_in[5];
    float* out = (float*)d_out;

    char* ws = (char*)d_ws;
    float*  route = (float*)ws;
    float*  s_in  = (float*)(ws + (size_t)T_TOK * E_EXP * 4);
    int8_t* q_in  = (int8_t*)(ws + (size_t)T_TOK * E_EXP * 4 + (size_t)T_TOK * 4);

    static bool attr_done = false;
    (void)attr_done;
    hipFuncSetAttribute((const void*)moe_kernel,
                        hipFuncAttributeMaxDynamicSharedMemorySize, SM_TOTAL);

    gate_kernel<<<T_TOK / 4, 128, 0, stream>>>(hs, gw, route);
    quant_kernel<<<T_TOK, 256, 0, stream>>>(hs, q_in, s_in);
    moe_kernel<<<T_TOK / 16, 256, SM_TOTAL, stream>>>(q_in, s_in, route,
                                                      w13, s13, w2, s2w, out);
}
